// Encoder_88656714924838
// MI455X (gfx1250) — matrix operand order repaired and checked by reading, not compiled, not run on hardware
//
#include <hip/hip_runtime.h>
#include <hip/hip_bf16.h>
#include <math.h>

// ---------------------------------------------------------------------------
// Graphormer-style encoder, fused per batch row for MI455X (gfx1250, wave32).
// One workgroup (256 threads = 8 wave32) per batch element; activations live
// in LDS for all 6 layers. GEMMs use v_wmma_f32_16x16x32_f16; the 64 MB MD
// distance-bias stream is DMA'd tile-by-tile into LDS by the Tensor Data
// Mover (tensor_load_to_lds + s_wait_tensorcnt), double-buffered per wave.
// ---------------------------------------------------------------------------

typedef __attribute__((ext_vector_type(16))) _Float16 v16h;
typedef __attribute__((ext_vector_type(8)))  float    v8f;
typedef __attribute__((ext_vector_type(4)))  unsigned int v4u;
typedef __attribute__((ext_vector_type(8)))  int      v8i;
typedef __attribute__((ext_vector_type(4)))  int      v4i;

#define LL   128   // sequence length
#define DM   64    // d_model
#define NH   4     // heads
#define DFF  512   // feed-forward dim
#define NLAY 6

__device__ __forceinline__ void wave_lds_fence() {
  // Intra-wave LDS write->read ordering (DS pipe is in-order per wave).
  asm volatile("s_wait_dscnt 0" ::: "memory");
}

__device__ __forceinline__ v8f wmma16(v16h a, v16h b, v8f c) {
  // D = A(16x32 f16) * B(32x16 f16) + C(16x16 f32)
  return __builtin_amdgcn_wmma_f32_16x16x32_f16(
      /*neg_a=*/false, a, /*neg_b=*/false, b,
      /*c_mod=*/(short)0, c, /*reuse_a=*/false, /*reuse_b=*/false);
}

// A fragment: 16x32, row m = lane&15; lanes 0-15 hold K 0..15,
// lanes 16-31 hold K 16..31 (contiguous, matches B fragments). 2x b128 loads.
__device__ __forceinline__ v16h load_a16(const _Float16* base, int stride, int lane) {
  const int m = lane & 15, hi = lane >> 4;
  const _Float16* r = base + m * stride + 16 * hi;
  v16h a;
#pragma unroll
  for (int j = 0; j < 8; ++j) a[j] = r[j];
#pragma unroll
  for (int j = 0; j < 8; ++j) a[8 + j] = r[8 + j];
  return a;
}

// B fragment: 32x16 row-major [K][N] from LDS f16.
__device__ __forceinline__ v16h load_b16(const _Float16* base, int stride, int lane) {
  const int n = lane & 15, koff = (lane >> 4) * 16;
  v16h b;
#pragma unroll
  for (int e = 0; e < 16; ++e) b[e] = base[(koff + e) * stride + n];
  return b;
}

// B fragment from global f32 weights (f16-converted on load; weights stay
// L2-resident across the 256 blocks: 1.9 MB total vs 192 MB L2).
__device__ __forceinline__ v16h load_b_w(const float* base, int stride, int lane) {
  const int n = lane & 15, koff = (lane >> 4) * 16;
  v16h b;
#pragma unroll
  for (int e = 0; e < 16; ++e) b[e] = (_Float16)base[(koff + e) * stride + n];
  return b;
}

__device__ __forceinline__ float half16_max(float v) {
  v = fmaxf(v, __shfl_xor(v, 1, 32));
  v = fmaxf(v, __shfl_xor(v, 2, 32));
  v = fmaxf(v, __shfl_xor(v, 4, 32));
  v = fmaxf(v, __shfl_xor(v, 8, 32));
  return v;
}
__device__ __forceinline__ float half16_sum(float v) {
  v += __shfl_xor(v, 1, 32);
  v += __shfl_xor(v, 2, 32);
  v += __shfl_xor(v, 4, 32);
  v += __shfl_xor(v, 8, 32);
  return v;
}

// TDM: DMA a 16x16 int32 tile (row stride 128 ints) from global into LDS.
// D# groups built per CDNA5 ISA §8: group0 = {flags,count | lds_addr |
// global_addr | type=2}; group1 = {data_size=4B, tensor dims (huge, OOB off),
// tile 16x16, dim0 stride 128}. Tracked by TENSORcnt, in-order per wave.
__device__ __forceinline__ void tdm_load_md16(unsigned int lds_byte, const int* gptr) {
  const unsigned long long ga = (unsigned long long)(uintptr_t)gptr;
  v4u g0;
  g0[0] = 1u;                                      // count=1 (valid descriptor)
  g0[1] = lds_byte;                                // lds_addr (bytes)
  g0[2] = (unsigned int)(ga & 0xffffffffu);        // global_addr[31:0]
  g0[3] = (unsigned int)((ga >> 32) & 0x1ffffffu)  // global_addr[56:32]
          | (2u << 30);                            // type=2 ("image")
  v8i g1;
  g1[0] = (int)(2u << 16);                         // data_size = 4 bytes
  g1[1] = 0;                                       // no atomic barrier
  g1[2] = 0x4000;                                  // tensor_dim0 = 1<<30
  g1[3] = (int)(0x10u | (16u << 16));              // tensor_dim1=1<<20, tile_dim0=16
  g1[4] = 16;                                      // tile_dim1 = 16 rows
  g1[5] = 128;                                     // tensor_dim0_stride = 128 ints
  g1[6] = 0;
  g1[7] = 0;
  v4i gz = {};
#if defined(__clang_major__) && (__clang_major__ >= 23)
  v8i gz8 = {};
  __builtin_amdgcn_tensor_load_to_lds(g0, g1, gz, gz, gz8, 0);
#else
  __builtin_amdgcn_tensor_load_to_lds(g0, g1, gz, gz, 0);
#endif
}

// x = LayerNorm(x + add), row-wise over DM; 128 threads each own a row.
__device__ __forceinline__ void ln_residual(float* xs, const float* add, int tid) {
  if (tid < LL) {
    const int r = tid * DM;
    float mu = 0.f;
#pragma unroll 8
    for (int d = 0; d < DM; ++d) mu += xs[r + d] + add[r + d];
    mu *= (1.f / (float)DM);
    float var = 0.f;
#pragma unroll 8
    for (int d = 0; d < DM; ++d) {
      float v = xs[r + d] + add[r + d] - mu;
      var += v * v;
    }
    var *= (1.f / (float)DM);
    const float inv = rsqrtf(var + 1e-5f);
#pragma unroll 8
    for (int d = 0; d < DM; ++d)
      xs[r + d] = (xs[r + d] + add[r + d] - mu) * inv;
  }
}

__global__ __launch_bounds__(256) void encoder_fused(
    const int* __restrict__ enc, const int* __restrict__ deg,
    const int* __restrict__ mdix,
    const float* __restrict__ semb, const float* __restrict__ demb,
    const float* __restrict__ memb,
    const float* __restrict__ Wq, const float* __restrict__ Wk,
    const float* __restrict__ Wv, const float* __restrict__ Wo,
    const float* __restrict__ W1, const float* __restrict__ W2,
    float* __restrict__ out) {
  __shared__ float    xs[LL * DM];        // 32 KB  persistent activations (f32)
  __shared__ _Float16 xh[LL * DM];        // 16 KB  f16 view of x / ctx buffer
  __shared__ _Float16 qs[LL * DM];        // 16 KB  Q  (FFN: per-wave relayout)
  __shared__ _Float16 ksm[LL * DM];       // 16 KB  K
  __shared__ _Float16 vsm[LL * DM];       // 16 KB  V
  __shared__ float    tmpf[LL * DM];      // 32 KB  attn scratch / residual tmp
  __shared__ int      mdbuf[8 * 2 * 256]; // 16 KB  per-wave MD tile dbl-buffer
  __shared__ float    membL[200 * NH];    // 3.2 KB md_emb table cache
  __shared__ float    maskb[LL];          // key pad-mask bias

  const int bb   = blockIdx.x;
  const int tid  = threadIdx.x;
  const int w    = tid >> 5;              // wave id 0..7
  const int lane = tid & 31;
  const int mrow = w * 16;                // this wave's M strip
  const int n16  = lane & 15;
  const int mb8  = (lane >> 4) * 8;       // D-fragment row base

  // ---- Phase 0: embeddings + positional encoding + caches ------------------
  for (int idx = tid; idx < LL * DM; idx += 256) {
    const int t = idx >> 6, d = idx & 63;
    const float div = __expf(-(float)(d & ~1) * (9.210340371976184f / 64.f));
    const float ang = (float)t * div;
    const float pe  = (d & 1) ? __cosf(ang) : __sinf(ang);
    const int e = enc[bb * LL + t];
    const int g = deg[bb * LL + t];
    xs[idx] = semb[e * DM + d] + demb[g * DM + d] + pe;
  }
  for (int i = tid; i < 200 * NH; i += 256) membL[i] = memb[i];
  if (tid < LL) maskb[tid] = (enc[bb * LL + tid] == 0) ? -1e9f : 0.f;
  __syncthreads();

  for (int layer = 0; layer < NLAY; ++layer) {
    const float* wq = Wq + layer * DM * DM;
    const float* wk = Wk + layer * DM * DM;
    const float* wv = Wv + layer * DM * DM;
    const float* wo = Wo + layer * DM * DM;
    const float* w1 = W1 + layer * DM * DFF;
    const float* w2 = W2 + layer * DFF * DM;

    // ---- x -> f16 ----------------------------------------------------------
    for (int idx = tid; idx < LL * DM; idx += 256) xh[idx] = (_Float16)xs[idx];
    __syncthreads();

    // ---- Q/K/V projections: each wave computes its 16-row strip -----------
    {
      const float* Wp[3] = {wq, wk, wv};
      _Float16*    Op[3] = {qs, ksm, vsm};
#pragma unroll
      for (int p = 0; p < 3; ++p) {
        for (int nt = 0; nt < 4; ++nt) {
          v8f acc = {};
#pragma unroll
          for (int kk = 0; kk < 2; ++kk) {
            v16h a = load_a16(xh + mrow * DM + 32 * kk, DM, lane);
            v16h b = load_b_w(Wp[p] + (32 * kk) * DM + nt * 16, DM, lane);
            acc = wmma16(a, b, acc);
          }
#pragma unroll
          for (int i = 0; i < 8; ++i)
            Op[p][(mrow + mb8 + i) * DM + nt * 16 + n16] = (_Float16)acc[i];
        }
      }
    }
    __syncthreads();

    // ---- Attention: 32 tasks (head, q-tile); 4 per wave --------------------
    {
      _Float16* attn = reinterpret_cast<_Float16*>(tmpf) + w * (16 * LL);
      const int*   mdt      = &mdbuf[w * 512];
      const unsigned int mdt_base =
          (unsigned int)(uintptr_t)&mdbuf[w * 512];   // LDS byte address

      for (int task = w; task < 32; task += 8) {
        const int h  = task & 3;
        const int qt = task >> 2;
        const int* md_row0 = mdix + (bb * LL + qt * 16) * LL;
        float sc[8][8];

        // Prefetch first MD bias-index tile via the Tensor Data Mover.
        tdm_load_md16(mdt_base, md_row0);

        // scores: S = (Q Khat^T) * 0.25 + bias + mask, via K-padded WMMA,
        // with the TDM streaming the next MD tile under the WMMA.
#pragma unroll
        for (int kt = 0; kt < 8; ++kt) {
          if (kt < 7)
            tdm_load_md16(mdt_base + ((kt + 1) & 1) * 1024, md_row0 + (kt + 1) * 16);

          v16h a = {};   // K-dim padded to 32: only K<16 populated
          {
            const int m = lane & 15, hi = lane >> 4;
#pragma unroll
            for (int j = 0; j < 16 * (1 - hi); ++j)
              a[j] = qs[(qt * 16 + m) * DM + h * 16 + j];
          }
          v16h b = {};
          if (lane < 16) {       // lanes 16-31 carry K=16..31 -> zero
#pragma unroll
            for (int e = 0; e < 16; ++e)
              b[e] = ksm[(kt * 16 + n16) * DM + h * 16 + e];
          }
          v8f z = {};
          v8f acc = wmma16(a, b, z);

          // Wait for tile kt (TDM ops complete in order; <=1 leaves only the
          // kt+1 prefetch outstanding).
          if (kt < 7) __builtin_amdgcn_s_wait_tensorcnt(1);
          else        __builtin_amdgcn_s_wait_tensorcnt(0);
          asm volatile("" ::: "memory");
          const int* tile = mdt + ((kt & 1) << 8);
#pragma unroll
          for (int i = 0; i < 8; ++i) {
            const int ki = kt * 16 + n16;
            const int dd = tile[(mb8 + i) * 16 + n16];
            sc[kt][i] = acc[i] * 0.25f + membL[dd * NH + h] + maskb[ki];
          }
        }
        // softmax over the 128 key columns of each row
#pragma unroll
        for (int i = 0; i < 8; ++i) {
          float mx = -3.4e38f;
#pragma unroll
          for (int kt = 0; kt < 8; ++kt) mx = fmaxf(mx, sc[kt][i]);
          mx = half16_max(mx);
          float sum = 0.f;
#pragma unroll
          for (int kt = 0; kt < 8; ++kt) {
            sc[kt][i] = __expf(sc[kt][i] - mx);
            sum += sc[kt][i];
          }
          sum = half16_sum(sum);
          const float inv = 1.f / sum;
#pragma unroll
          for (int kt = 0; kt < 8; ++kt)
            attn[(mb8 + i) * LL + kt * 16 + n16] = (_Float16)(sc[kt][i] * inv);
        }
        wave_lds_fence();
        // ctx(16x16) = attn(16x128) @ V[:, h*16 : h*16+16]
        v8f cacc = {};
#pragma unroll
        for (int st = 0; st < 4; ++st) {
          v16h a = load_a16(attn + 32 * st, LL, lane);
          v16h b = load_b16(vsm + (32 * st) * DM + h * 16, DM, lane);
          cacc = wmma16(a, b, cacc);
        }
#pragma unroll
        for (int i = 0; i < 8; ++i)
          xh[(qt * 16 + mb8 + i) * DM + h * 16 + n16] = (_Float16)cacc[i];
        wave_lds_fence();
      }
    }
    __syncthreads();

    // ---- Output projection: o = ctx @ Wo  (each wave writes its own strip,
    //      which exactly aliases its own already-consumed attn scratch) ------
    {
#pragma unroll
      for (int nt = 0; nt < 4; ++nt) {
        v8f acc = {};
#pragma unroll
        for (int kk = 0; kk < 2; ++kk) {
          v16h a = load_a16(xh + mrow * DM + 32 * kk, DM, lane);
          v16h b = load_b_w(wo + (32 * kk) * DM + nt * 16, DM, lane);
          acc = wmma16(a, b, acc);
        }
#pragma unroll
        for (int i = 0; i < 8; ++i)
          tmpf[(mrow + mb8 + i) * DM + nt * 16 + n16] = acc[i];
      }
    }
    __syncthreads();
    ln_residual(xs, tmpf, tid);     // x = LN(o + x)
    __syncthreads();

    // ---- FFN: chunk FF dim in 32-wide slices, hidden never materializes ----
    for (int idx = tid; idx < LL * DM; idx += 256) xh[idx] = (_Float16)xs[idx];
    __syncthreads();
    {
      _Float16* rb = qs + w * (16 * 32);   // per-wave D->A relayout buffer
      v8f facc[4];
#pragma unroll
      for (int nt = 0; nt < 4; ++nt) { v8f z = {}; facc[nt] = z; }
      for (int c = 0; c < 16; ++c) {
        if (c < 15) {   // pull next weight chunk toward L2/L0 early
          __builtin_prefetch(w1 + (c + 1) * 32, 0, 0);
          __builtin_prefetch(w2 + ((c + 1) * 32) * DM, 0, 0);
        }
#pragma unroll
        for (int nt2 = 0; nt2 < 2; ++nt2) {
          v8f hacc = {};
#pragma unroll
          for (int kk = 0; kk < 2; ++kk) {
            v16h a = load_a16(xh + mrow * DM + 32 * kk, DM, lane);
            v16h b = load_b_w(w1 + (32 * kk) * DFF + c * 32 + nt2 * 16, DFF, lane);
            hacc = wmma16(a, b, hacc);
          }
#pragma unroll
          for (int i = 0; i < 8; ++i)
            rb[(mb8 + i) * 32 + nt2 * 16 + n16] = (_Float16)fmaxf(hacc[i], 0.f);
        }
        wave_lds_fence();
#pragma unroll
        for (int nt = 0; nt < 4; ++nt) {
          v16h a = load_a16(rb, 32, lane);
          v16h b = load_b_w(w2 + (c * 32) * DM + nt * 16, DM, lane);
          facc[nt] = wmma16(a, b, facc[nt]);
        }
        wave_lds_fence();   // rb read done before next chunk overwrites it
      }
#pragma unroll
      for (int nt = 0; nt < 4; ++nt)
#pragma unroll
        for (int i = 0; i < 8; ++i)
          tmpf[(mrow + mb8 + i) * DM + nt * 16 + n16] = facc[nt][i];
    }
    __syncthreads();
    ln_residual(xs, tmpf, tid);     // x = LN(ffn + x)
    __syncthreads();
  }

  // ---- Write result --------------------------------------------------------
  for (int idx = tid; idx < LL * DM; idx += 256)
    out[(long)bb * (LL * DM) + idx] = xs[idx];
}

extern "C" void kernel_launch(void* const* d_in, const int* in_sizes, int n_in,
                              void* d_out, int out_size, void* d_ws, size_t ws_size,
                              hipStream_t stream) {
  (void)in_sizes; (void)n_in; (void)out_size; (void)d_ws; (void)ws_size;
  const int*   enc  = (const int*)d_in[0];
  const int*   deg  = (const int*)d_in[1];
  const int*   mdix = (const int*)d_in[2];
  const float* semb = (const float*)d_in[3];
  const float* demb = (const float*)d_in[4];
  const float* memb = (const float*)d_in[5];
  const float* Wq   = (const float*)d_in[6];
  const float* Wk   = (const float*)d_in[7];
  const float* Wv   = (const float*)d_in[8];
  const float* Wo   = (const float*)d_in[9];
  const float* W1   = (const float*)d_in[10];
  const float* W2   = (const float*)d_in[11];
  float*       out  = (float*)d_out;

  encoder_fused<<<dim3(256), dim3(256), 0, stream>>>(
      enc, deg, mdix, semb, demb, memb, Wq, Wk, Wv, Wo, W1, W2, out);
}
